// _PointnetSAModuleBase_16561393893688
// MI455X (gfx1250) — matrix operand order repaired and checked by reading, not compiled, not run on hardware
//
#include <hip/hip_runtime.h>
#include <hip/hip_bf16.h>

typedef __attribute__((ext_vector_type(16))) _Float16 v16h;
typedef __attribute__((ext_vector_type(8)))  _Float16 v8h;
typedef __attribute__((ext_vector_type(8)))  float    v8f;

#define NPTS    8192
#define NBATCH  8
#define NPT     2048
#define NS      32
#define CIN     64
#define K1      67     // 3 + 64
#define K1P     96     // padded to 3 k-tiles of 32
#define H1      64
#define H2      128
#define RAD2    0.04f

// ---------------------------------------------------------------------------
// Kernel 1: furthest point sampling. One block (1024 threads) per batch.
// Each thread owns 8 points + running min-distance entirely in registers.
// ---------------------------------------------------------------------------
__global__ __launch_bounds__(1024) void fps_kernel(
    const float* __restrict__ xyz, int* __restrict__ fps_idx,
    float* __restrict__ out_newxyz, float* __restrict__ out_fpsidx)
{
    __shared__ float wbest[32];
    __shared__ int   widx[32];
    __shared__ int   s_last;
    __shared__ int   s_idx[NPT];

    const int b   = blockIdx.x;
    const int tid = threadIdx.x;
    const float* base = xyz + (size_t)b * NPTS * 3;

    float px[8], py[8], pz[8], dist[8];
#pragma unroll
    for (int k = 0; k < 8; ++k) {
        int i = tid + k * 1024;
        px[k] = base[3 * i + 0];
        py[k] = base[3 * i + 1];
        pz[k] = base[3 * i + 2];
        dist[k] = 1e10f;
    }

    if (tid == 0) s_idx[0] = 0;
    int last = 0;

    for (int s = 1; s < NPT; ++s) {
        float lx = base[3 * last + 0];
        float ly = base[3 * last + 1];
        float lz = base[3 * last + 2];

        float best = -1.0f;
        int   bi   = 0;
#pragma unroll
        for (int k = 0; k < 8; ++k) {
            float dx = px[k] - lx, dy = py[k] - ly, dz = pz[k] - lz;
            float d2 = dx * dx + dy * dy + dz * dz;
            float d  = fminf(dist[k], d2);
            dist[k]  = d;
            if (d > best) { best = d; bi = tid + k * 1024; }
        }
        // wave32 reduction (argmax, low index on tie)
#pragma unroll
        for (int off = 16; off; off >>= 1) {
            float ob = __shfl_xor(best, off, 32);
            int   oi = __shfl_xor(bi,   off, 32);
            if (ob > best || (ob == best && oi < bi)) { best = ob; bi = oi; }
        }
        if ((tid & 31) == 0) { wbest[tid >> 5] = best; widx[tid >> 5] = bi; }
        __syncthreads();
        if (tid < 32) {
            float bb = wbest[tid];
            int   ii = widx[tid];
#pragma unroll
            for (int off = 16; off; off >>= 1) {
                float ob = __shfl_xor(bb, off, 32);
                int   oi = __shfl_xor(ii, off, 32);
                if (ob > bb || (ob == bb && oi < ii)) { bb = ob; ii = oi; }
            }
            if (tid == 0) { s_last = ii; s_idx[s] = ii; }
        }
        __syncthreads();
        last = s_last;
    }

    // Emit fps_idx (int ws + float out) and new_xyz
    for (int p = tid; p < NPT; p += 1024) {
        int idx = s_idx[p];
        fps_idx[b * NPT + p] = idx;
        out_fpsidx[b * NPT + p] = (float)idx;
        size_t o = (size_t)(b * NPT + p) * 3;
        out_newxyz[o + 0] = base[3 * idx + 0];
        out_newxyz[o + 1] = base[3 * idx + 1];
        out_newxyz[o + 2] = base[3 * idx + 2];
    }
}

// ---------------------------------------------------------------------------
// Kernel 2: ball query. One wave32 per centroid; ballot-compacted scan gives
// the first NS in-radius indices in ascending order (matches reference sort).
// ---------------------------------------------------------------------------
__global__ __launch_bounds__(256) void ballq_kernel(
    const float* __restrict__ xyz, const float* __restrict__ newxyz,
    int* __restrict__ gidx)
{
    const int lane = threadIdx.x & 31;
    const int wave = blockIdx.x * (blockDim.x >> 5) + (threadIdx.x >> 5);
    const int b = wave >> 11;          // / NPT
    const int p = wave & (NPT - 1);

    const float* nx = newxyz + (size_t)(b * NPT + p) * 3;
    const float cx = nx[0], cy = nx[1], cz = nx[2];
    const float* base = xyz + (size_t)b * NPTS * 3;
    int* out = gidx + (size_t)(b * NPT + p) * NS;

    int cnt = 0;
    int first = -1;
    for (int start = 0; start < NPTS; start += 32) {
        int i = start + lane;
        float dx = base[3 * i + 0] - cx;
        float dy = base[3 * i + 1] - cy;
        float dz = base[3 * i + 2] - cz;
        float d2 = dx * dx + dy * dy + dz * dz;
        bool hit = (d2 <= RAD2);
        unsigned mask = (unsigned)__ballot(hit);
        int pos = cnt + __popc(mask & ((1u << lane) - 1u));
        if (hit && pos < NS) out[pos] = i;
        if (first < 0 && mask != 0u) first = start + (__ffs(mask) - 1);
        cnt += __popc(mask);
        if (cnt >= NS) break;
    }
    if (first < 0) first = 0;
    if (cnt > NS) cnt = NS;
    for (int j = cnt + lane; j < NS; j += 32) out[j] = first;
}

// ---------------------------------------------------------------------------
// Kernel 3: gather + 2-layer MLP (WMMA f16 -> f32) + max-pool over samples.
// One wave per centroid. Weights live K-contiguous (transposed) in LDS so
// every B fragment is a single 32B contiguous read (ds_load_b128 pair).
// ---------------------------------------------------------------------------
__device__ __forceinline__ v8f wmma_f16(v16h a, v16h b, v8f c) {
    return __builtin_amdgcn_wmma_f32_16x16x32_f16(false, a, false, b,
                                                  (short)0, c, false, false);
}

__global__ __launch_bounds__(256) void mlp_kernel(
    const float* __restrict__ xyz, const float* __restrict__ feat,
    const float* __restrict__ W1, const float* __restrict__ b1,
    const float* __restrict__ W2, const float* __restrict__ b2,
    const float* __restrict__ newxyz, const int* __restrict__ gidx,
    float* __restrict__ out_feat)
{
    // Transposed (n-major, K-contiguous) f16 weights
    __shared__ __align__(32) _Float16 sW1t[H1 * K1P];   // 12 KB
    __shared__ __align__(32) _Float16 sW2t[H2 * H1];    // 16 KB
    __shared__ __align__(32) _Float16 sH[8][NS * H1];   // 32 KB

    const int tid = threadIdx.x;

    // Stage weights as f16, transposed so K is the fast axis
    for (int i = tid; i < H1 * K1P; i += 256) {
        int n = i / K1P;
        int k = i - n * K1P;
        sW1t[i] = (k < K1) ? (_Float16)W1[k * H1 + n] : (_Float16)0.0f;
    }
    for (int i = tid; i < H2 * H1; i += 256) {
        int n = i >> 6;
        int k = i & 63;
        sW2t[i] = (_Float16)W2[k * H2 + n];
    }
    __syncthreads();

    const int lane  = tid & 31;
    const int w     = tid >> 5;
    const int wave  = blockIdx.x * 8 + w;
    const int b     = wave >> 11;
    const int p     = wave & (NPT - 1);
    const int col16 = lane & 15;
    const int h     = lane >> 4;

    const int*   gi = gidx + (size_t)(b * NPT + p) * NS;
    const float* nx = newxyz + (size_t)(b * NPT + p) * 3;
    const float  cx = nx[0], cy = nx[1], cz = nx[2];

    // ---- Build layer-1 A fragments straight from the gather --------------
    // A layout chosen here (contiguous k, same as B): lane holds row (lane&15); element e -> K =
    //   32*kt + 16*(lane>>4) + e
    v16h a1[2][3];
#pragma unroll
    for (int mt = 0; mt < 2; ++mt) {
        int s   = mt * 16 + col16;
        int idx = gi[s];
        const float* prow = xyz  + ((size_t)b * NPTS + idx) * 3;
        const float* frow = feat + ((size_t)b * NPTS + idx) * CIN;
        float g0 = prow[0] - cx, g1 = prow[1] - cy, g2 = prow[2] - cz;
#pragma unroll
        for (int kt = 0; kt < 3; ++kt) {
            v16h av;
#pragma unroll
            for (int e = 0; e < 16; ++e) {
                int c = kt * 32 + h * 16 + e;
                float v;
                if      (c == 0) v = g0;
                else if (c == 1) v = g1;
                else if (c == 2) v = g2;
                else if (c < K1) v = frow[c - 3];
                else             v = 0.0f;
                av[e] = (_Float16)v;
            }
            a1[mt][kt] = av;
        }
    }

    // ---- Layer 1: (32x96) x (96x64) --------------------------------------
    v8f acc[2][4];
#pragma unroll
    for (int mt = 0; mt < 2; ++mt)
#pragma unroll
        for (int nt = 0; nt < 4; ++nt)
            acc[mt][nt] = (v8f){};

#pragma unroll
    for (int nt = 0; nt < 4; ++nt) {
        int col = nt * 16 + col16;
#pragma unroll
        for (int kt = 0; kt < 3; ++kt) {
            // B layout: lane = column; element e -> K = 32*kt + 16*h + e,
            // contiguous in the transposed weight tile.
            v16h bv = *(const v16h*)(sW1t + col * K1P + kt * 32 + h * 16);
            acc[0][nt] = wmma_f16(a1[0][kt], bv, acc[0][nt]);
            acc[1][nt] = wmma_f16(a1[1][kt], bv, acc[1][nt]);
        }
    }

    // Bias + ReLU, spill to LDS row-major f16 (D layout: VGPR r -> row
    // 16*mt + 8*h + r, column lane&15 within the n-tile).
#pragma unroll
    for (int nt = 0; nt < 4; ++nt) {
        int col = nt * 16 + col16;
        float bias = b1[col];
#pragma unroll
        for (int mt = 0; mt < 2; ++mt) {
#pragma unroll
            for (int r = 0; r < 8; ++r) {
                float v = fmaxf(acc[mt][nt][r] + bias, 0.0f);
                int row = mt * 16 + h * 8 + r;
                sH[w][row * H1 + col] = (_Float16)v;
            }
        }
    }
    __syncthreads();

    // ---- Layer 2: (32x64) x (64x128), fused max-pool epilogue ------------
    // A fragments: two 16B-contiguous runs of 8 K-values per lane.
    v16h a2[2][2];
#pragma unroll
    for (int mt = 0; mt < 2; ++mt) {
        int row = mt * 16 + col16;
#pragma unroll
        for (int kt = 0; kt < 2; ++kt) {
            const _Float16* pbase = &sH[w][row * H1 + kt * 32 + h * 16];
            v8h lo = *(const v8h*)(pbase);
            v8h hi = *(const v8h*)(pbase + 8);
            a2[mt][kt] = __builtin_shufflevector(lo, hi,
                0, 1, 2, 3, 4, 5, 6, 7, 8, 9, 10, 11, 12, 13, 14, 15);
        }
    }

#pragma unroll
    for (int nt = 0; nt < 8; ++nt) {
        int col = nt * 16 + col16;
        v8f c0 = (v8f){};
        v8f c1 = (v8f){};
#pragma unroll
        for (int kt = 0; kt < 2; ++kt) {
            v16h bv = *(const v16h*)(sW2t + col * H1 + kt * 32 + h * 16);
            c0 = wmma_f16(a2[0][kt], bv, c0);
            c1 = wmma_f16(a2[1][kt], bv, c1);
        }
        float bias = b2[col];
        float m = 0.0f;                        // ReLU outputs are >= 0
#pragma unroll
        for (int r = 0; r < 8; ++r) {
            m = fmaxf(m, fmaxf(c0[r] + bias, 0.0f));
            m = fmaxf(m, fmaxf(c1[r] + bias, 0.0f));
        }
        m = fmaxf(m, __shfl_xor(m, 16, 32));   // merge the two row halves
        if (h == 0)
            out_feat[((size_t)(b * H2 + col)) * NPT + p] = m;
    }
}

// ---------------------------------------------------------------------------
extern "C" void kernel_launch(void* const* d_in, const int* in_sizes, int n_in,
                              void* d_out, int out_size, void* d_ws, size_t ws_size,
                              hipStream_t stream) {
    const float* xyz  = (const float*)d_in[0];
    const float* feat = (const float*)d_in[1];
    const float* W1   = (const float*)d_in[2];
    const float* b1   = (const float*)d_in[3];
    const float* W2   = (const float*)d_in[4];
    const float* b2   = (const float*)d_in[5];

    float* out_newxyz = (float*)d_out;                         // B*NPT*3
    float* out_feat   = out_newxyz + (size_t)NBATCH * NPT * 3; // B*H2*NPT
    float* out_fpsidx = out_feat + (size_t)NBATCH * H2 * NPT;  // B*NPT

    int* ws_fpsidx = (int*)d_ws;                               // B*NPT
    int* ws_gidx   = ws_fpsidx + NBATCH * NPT;                 // B*NPT*NS

    fps_kernel<<<dim3(NBATCH), dim3(1024), 0, stream>>>(
        xyz, ws_fpsidx, out_newxyz, out_fpsidx);

    ballq_kernel<<<dim3(NBATCH * NPT * 32 / 256), dim3(256), 0, stream>>>(
        xyz, out_newxyz, ws_gidx);

    mlp_kernel<<<dim3(NBATCH * NPT / 8), dim3(256), 0, stream>>>(
        xyz, feat, W1, b1, W2, b2, out_newxyz, ws_gidx, out_feat);
}
